// SAGE_37873021616716
// MI455X (gfx1250) — compile-verified, matrix operand order checked by reading, not run on hardware
//
#include <hip/hip_runtime.h>

// Problem constants (fixed by the reference)
#define N_NODES 20000
#define T_DIM   32
#define E_EDGES 640000
#define L_LAYERS 2
#define C_OUT   10
#define K_W     5
#define PAD_W   2

typedef __attribute__((ext_vector_type(16))) _Float16 v16h;
typedef __attribute__((ext_vector_type(8)))  float    v8f;

// ---------------- zero workspace region ----------------
__global__ void zero_kernel(float* __restrict__ p, int n) {
    int i = blockIdx.x * blockDim.x + threadIdx.x;
    if (i < n) p[i] = 0.0f;
}

// ---------------- degree: cnt[dst[e]] += 1 ----------------
__global__ void degree_kernel(const int* __restrict__ dst, float* __restrict__ cnt, int e) {
    int i = blockIdx.x * blockDim.x + threadIdx.x;
    if (i < e) unsafeAtomicAdd(&cnt[dst[i]], 1.0f);
}

// ---------------- per-column sum / sumsq over N rows ----------------
__global__ void stats_kernel(const float* __restrict__ x, float* __restrict__ s12, int n) {
    __shared__ float sh1[256];
    __shared__ float sh2[256];
    int t = blockIdx.x;                  // one block per column t (32 blocks)
    float s1 = 0.f, s2 = 0.f;
    for (int i = threadIdx.x; i < n; i += blockDim.x) {
        float v = x[i * T_DIM + t];
        s1 += v; s2 += v * v;
    }
    sh1[threadIdx.x] = s1; sh2[threadIdx.x] = s2;
    __syncthreads();
    for (int o = 128; o > 0; o >>= 1) {
        if (threadIdx.x < o) { sh1[threadIdx.x] += sh1[threadIdx.x + o];
                               sh2[threadIdx.x] += sh2[threadIdx.x + o]; }
        __syncthreads();
    }
    if (threadIdx.x == 0) { s12[t] = sh1[0]; s12[T_DIM + t] = sh2[0]; }
}

// ---------------- GraphNorm: xn = (x - a*m)*sqrtN/||x-a*m|| * scale + shift; also zero aggr ----------------
__global__ void norm_kernel(const float* __restrict__ x, const float* __restrict__ s12,
                            const float* __restrict__ alpha, const float* __restrict__ scale,
                            const float* __restrict__ shift,
                            float* __restrict__ xn, float* __restrict__ aggr, int total) {
    int i = blockIdx.x * blockDim.x + threadIdx.x;
    if (i >= total) return;
    int t = i & (T_DIM - 1);
    float s1 = s12[t], s2 = s12[T_DIM + t];
    float a  = alpha[t];
    float m  = s1 * (1.0f / (float)N_NODES);
    float am = a * m;
    float nrm2 = s2 - 2.0f * am * s1 + (float)N_NODES * am * am;
    float inv  = sqrtf((float)N_NODES) * rsqrtf(nrm2);
    xn[i]   = (x[i] - am) * inv * scale[t] + shift[t];
    aggr[i] = 0.0f;
}

// ---------------- per-node Conv1d(1->C, K=5, 'same') + bias + max over channels, via WMMA ----------------
// One wave per node.  A = 16x32 f16 im2col patches (taps 0..4, rest zero),
// B = 32x16 f16 weights (col c = lane&15, element e = tap e), C/D = 16x16 f32.
__global__ void conv_wmma_kernel(const float* __restrict__ xn,
                                 const float* __restrict__ W,   // [C_OUT, K_W] for this layer
                                 const float* __restrict__ bias,// [C_OUT]
                                 float* __restrict__ y, int n_nodes) {
    int wave = (blockIdx.x * blockDim.x + threadIdx.x) >> 5;   // wave-uniform
    int lane = threadIdx.x & 31;
    if (wave >= n_nodes) return;                               // whole-wave exit: EXEC stays all-1 for WMMA
    const int n = wave;
    const int col = lane & 15;

    // Build B once per wave: lanes 0..15 hold K=0..15 (element e = tap e), lanes 16..31 hold K>=16 (all zero taps)
    v16h b;
#pragma unroll
    for (int q = 0; q < 16; ++q) b[q] = (_Float16)0.0f;
    if (lane < 16 && col < C_OUT) {
#pragma unroll
        for (int k = 0; k < K_W; ++k) b[k] = (_Float16)W[col * K_W + k];
    }
    const bool  colvalid = (col < C_OUT);
    const float biasv    = colvalid ? bias[col] : 0.0f;
    const float* row = xn + (size_t)n * T_DIM;

#pragma unroll
    for (int tile = 0; tile < 2; ++tile) {
        const int t0 = tile * 16;
        // A: lanes 0..15 hold row t=t0+lane, elements 0..4 = taps; lanes 16..31 (K=8..15,24..31) all zero
        v16h a;
#pragma unroll
        for (int q = 0; q < 16; ++q) a[q] = (_Float16)0.0f;
        if (lane < 16) {
            const int t = t0 + lane;
#pragma unroll
            for (int k = 0; k < K_W; ++k) {
                const int idx = t + k - PAD_W;
                float v = (idx >= 0 && idx < T_DIM) ? row[idx] : 0.0f;
                a[k] = (_Float16)v;
            }
        }
        v8f c = {};
        c = __builtin_amdgcn_wmma_f32_16x16x32_f16(
                /*neg_a=*/false, a, /*neg_b=*/false, b,
                /*c_mod=*/(short)0, c, /*reuse_a=*/false, /*reuse_b=*/false);

        // c[v]: row t = t0 + v + 8*(lane>=16), col = lane&15.  Max over channels = butterfly within 16-lane half.
        float m[8];
#pragma unroll
        for (int v = 0; v < 8; ++v) {
            float val = colvalid ? (c[v] + biasv) : -3.0e38f;
#pragma unroll
            for (int off = 1; off < 16; off <<= 1)
                val = fmaxf(val, __shfl_xor(val, off, 32));
            m[v] = val;
        }
        if (col == 0) {
            const int base = n * T_DIM + t0 + ((lane >> 4) ? 8 : 0);
#pragma unroll
            for (int v = 0; v < 8; ++v) y[base + v] = m[v];
        }
    }
}

// ---------------- scatter-add: aggr[dst[e], t] += y[src[e], t] ----------------
__global__ void scatter_kernel(const float* __restrict__ y,
                               const int* __restrict__ src, const int* __restrict__ dst,
                               float* __restrict__ aggr, int total) {
    int i = blockIdx.x * blockDim.x + threadIdx.x;
    if (i >= total) return;
    int e = i >> 5;
    int t = i & 31;
    unsafeAtomicAdd(&aggr[dst[e] * T_DIM + t], y[src[e] * T_DIM + t]);
}

// ---------------- update: x = relu(0.5*(xn + aggr/denom)) ----------------
__global__ void update_kernel(const float* __restrict__ xn, const float* __restrict__ aggr,
                              const float* __restrict__ cnt, float* __restrict__ xout, int total) {
    int i = blockIdx.x * blockDim.x + threadIdx.x;
    if (i >= total) return;
    int nd = i >> 5;
    float d = fmaxf(cnt[nd], 1.0f);
    float v = 0.5f * (xn[i] + aggr[i] / d);
    xout[i] = fmaxf(v, 0.0f);
}

// ---------------- per-node temporal sum: s[n] = sum_t x[n,t] (one wave per node) ----------------
__global__ void rowsum_kernel(const float* __restrict__ x, float* __restrict__ s, int total) {
    int i = blockIdx.x * blockDim.x + threadIdx.x;
    if (i >= total) return;
    float v = x[i];
#pragma unroll
    for (int o = 16; o > 0; o >>= 1) v += __shfl_xor(v, o, 32);
    if ((threadIdx.x & 31) == 0) s[i >> 5] = v;
}

// ---------------- out[j] = Wout[j,:] . s + bout[j], one block per output ----------------
__global__ void out_kernel(const float* __restrict__ W, const float* __restrict__ bout,
                           const float* __restrict__ s, float* __restrict__ out, int n) {
    __shared__ float sh[256];
    int j = blockIdx.x;
    float acc = 0.0f;
    for (int i = threadIdx.x; i < n; i += blockDim.x) acc += W[j * n + i] * s[i];
    sh[threadIdx.x] = acc;
    __syncthreads();
    for (int o = 128; o > 0; o >>= 1) {
        if (threadIdx.x < o) sh[threadIdx.x] += sh[threadIdx.x + o];
        __syncthreads();
    }
    if (threadIdx.x == 0) out[j] = sh[0] + bout[j];
}

extern "C" void kernel_launch(void* const* d_in, const int* in_sizes, int n_in,
                              void* d_out, int out_size, void* d_ws, size_t ws_size,
                              hipStream_t stream) {
    const float* x_in  = (const float*)d_in[0];   // [N, T]
    const float* convW = (const float*)d_in[1];   // [L, C, 1, K]
    const float* convb = (const float*)d_in[2];   // [L, C]
    const float* alpha = (const float*)d_in[3];   // [L, T]
    const float* scale = (const float*)d_in[4];   // [L, T]
    const float* shift = (const float*)d_in[5];   // [L, T]
    const float* Wout  = (const float*)d_in[6];   // [3, N]
    const float* bout  = (const float*)d_in[7];   // [3]
    const int*   ei    = (const int*)d_in[8];     // [2, E]
    const int* src = ei;
    const int* dst = ei + E_EDGES;
    float* out = (float*)d_out;

    // workspace layout (floats)
    float* ws   = (float*)d_ws;
    float* cnt  = ws;                              // N
    float* xbuf = cnt  + N_NODES;                  // N*T (x after each layer)
    float* xn   = xbuf + N_NODES * T_DIM;          // N*T (normalized x)
    float* ybuf = xn   + N_NODES * T_DIM;          // N*T (per-node conv+max)
    float* aggr = ybuf + N_NODES * T_DIM;          // N*T (scatter target)
    float* s12  = aggr + N_NODES * T_DIM;          // 2*T
    float* svec = s12  + 2 * T_DIM;                // N

    const int NT = N_NODES * T_DIM;
    const int ET = E_EDGES * T_DIM;
    const int TB = 256;

    // degree
    zero_kernel<<<(N_NODES + TB - 1) / TB, TB, 0, stream>>>(cnt, N_NODES);
    degree_kernel<<<(E_EDGES + TB - 1) / TB, TB, 0, stream>>>(dst, cnt, E_EDGES);

    const float* xcur = x_in;
    for (int l = 0; l < L_LAYERS; ++l) {
        stats_kernel<<<T_DIM, TB, 0, stream>>>(xcur, s12, N_NODES);
        norm_kernel<<<(NT + TB - 1) / TB, TB, 0, stream>>>(
            xcur, s12, alpha + l * T_DIM, scale + l * T_DIM, shift + l * T_DIM,
            xn, aggr, NT);
        // one wave per node, 8 waves per 256-thread block
        conv_wmma_kernel<<<(N_NODES + 7) / 8, TB, 0, stream>>>(
            xn, convW + l * C_OUT * K_W, convb + l * C_OUT, ybuf, N_NODES);
        scatter_kernel<<<(ET + TB - 1) / TB, TB, 0, stream>>>(ybuf, src, dst, aggr, ET);
        update_kernel<<<(NT + TB - 1) / TB, TB, 0, stream>>>(xn, aggr, cnt, xbuf, NT);
        xcur = xbuf;
    }

    rowsum_kernel<<<(NT + TB - 1) / TB, TB, 0, stream>>>(xbuf, svec, NT);
    out_kernel<<<3, TB, 0, stream>>>(Wout, bout, svec, out, N_NODES);
}
